// CrossAttention_8358006358080
// MI455X (gfx1250) — matrix operand order repaired and checked by reading, not compiled, not run on hardware
//
#include <hip/hip_runtime.h>

// ---------------------------------------------------------------------------
// CDNA5 (gfx1250) wave32 + WMMA bf16 implementation of cross-attention.
// q=x@Wq; k,v=ctx@Wk/Wv; flash-attention softmax(qk^T*0.125)v; out@Wo+bo.
// Matrix math via v_wmma_f32_16x16x32_bf16 (fp32 accumulate).
// bf16 tiles staged with global_load_async_to_lds_b128 (ASYNCcnt),
// softmax reductions via DPP row_xmask butterflies (VALU, not LDS pipe).
// ---------------------------------------------------------------------------

typedef __attribute__((ext_vector_type(16))) __bf16       v16bf;
typedef __attribute__((ext_vector_type(8)))  float        v8f;
typedef __attribute__((ext_vector_type(4)))  unsigned int u32x4;
typedef __attribute__((ext_vector_type(4)))  float        f32x4;

union ABfrag { u32x4 u[2]; v16bf v; };

__device__ __forceinline__ unsigned short f2bf(float f) {
  __bf16 h = (__bf16)f;  // RNE, native cvt on gfx1250
  return __builtin_bit_cast(unsigned short, h);
}

#define WMMA_BF16(a, b, c) \
  __builtin_amdgcn_wmma_f32_16x16x32_bf16(false, (a), false, (b), (short)0, (c), false, false)

// DPP16 lane permute; CTRL must be an ICE -> template parameter.
template <int CTRL>
__device__ __forceinline__ float dppf(float x) {
  return __builtin_bit_cast(
      float, __builtin_amdgcn_update_dpp(0, __builtin_bit_cast(int, x), CTRL,
                                         0xF, 0xF, true));
}
// Butterfly reductions across each 16-lane row (row_xmask = 0x160|mask).
__device__ __forceinline__ float rowmax16(float m) {
  m = fmaxf(m, dppf<0x161>(m));
  m = fmaxf(m, dppf<0x162>(m));
  m = fmaxf(m, dppf<0x164>(m));
  m = fmaxf(m, dppf<0x168>(m));
  return m;
}
__device__ __forceinline__ float rowsum16(float s) {
  s += dppf<0x161>(s);
  s += dppf<0x162>(s);
  s += dppf<0x164>(s);
  s += dppf<0x168>(s);
  return s;
}

// ---------------------------------------------------------------------------
// GEMM: C[M,N] = A[M,K] @ B[K,N] (+bias).  A fp32 or bf16 (-> bf16),
// B fp32 (-> bf16 on stage-in), C bf16 or fp32.
// Block = 256 threads = 8 waves; tile 128x128, K-step 32 (one WMMA K).
// Wave grid 4(M)x2(N); each wave owns a 32x64 tile = 2x4 WMMA accumulators.
// ---------------------------------------------------------------------------
template <bool A_BF16, bool OUT_F32>
__global__ __launch_bounds__(256) void gemm_bf16_wmma(
    const void* __restrict__ Ap, const float* __restrict__ B,
    void* __restrict__ Cp, const float* __restrict__ bias,
    int M, int N, int K) {
  __shared__ alignas(16) unsigned short As[128][32];  // [m][k]
  __shared__ alignas(16) unsigned short Bt[128][32];  // [n][k]  (transposed!)

  const int tid  = threadIdx.x;
  const int wave = tid >> 5, lane = tid & 31;
  const int wm = wave >> 1, wn = wave & 1;
  const int lrow = lane & 15, lhi = lane >> 4;
  const int koffA = lhi * 16;   // A-frag half-wave K offset (contiguous k = 16*lhi + s)
  const int koffB = lhi * 16;   // B-frag half-wave K offset (contiguous k = 16*lhi + s)
  const int bm = blockIdx.y, bn = blockIdx.x;

  v8f acc[2][4];
  {
    v8f z = {};
#pragma unroll
    for (int i = 0; i < 2; i++)
#pragma unroll
      for (int j = 0; j < 4; j++) acc[i][j] = z;
  }

  const int arow = tid >> 1, acol0 = (tid & 1) * 16;  // 128 rows x 32 k
  const int brow = tid >> 3, bcol0 = (tid & 7) * 16;  // 32 k x 128 n

  for (int k0 = 0; k0 < K; k0 += 32) {
    // ---- stage A tile ----
    {
      size_t base = (size_t)(bm * 128 + arow) * K + k0 + acol0;
      if constexpr (A_BF16) {
        // bf16 already: async DMA straight into LDS (no VGPR round trip)
        unsigned lds = (unsigned)(uintptr_t)&As[arow][acol0];
        const unsigned short* gp = (const unsigned short*)Ap + base;
        asm volatile("global_load_async_to_lds_b128 %0, %1, off" ::"v"(lds),
                     "v"(gp)
                     : "memory");
        asm volatile(
            "global_load_async_to_lds_b128 %0, %1, off offset:16" ::"v"(lds),
            "v"(gp)
            : "memory");
        if (k0 + 32 < K)
          __builtin_prefetch((const unsigned short*)Ap + base + 32, 0, 1);
      } else {
        const f32x4* p = (const f32x4*)((const float*)Ap + base);
        alignas(16) unsigned short tmp[16];
#pragma unroll
        for (int i = 0; i < 4; i++) {
          f32x4 f = p[i];
          tmp[i * 4 + 0] = f2bf(f.x);
          tmp[i * 4 + 1] = f2bf(f.y);
          tmp[i * 4 + 2] = f2bf(f.z);
          tmp[i * 4 + 3] = f2bf(f.w);
        }
        *(u32x4*)&As[arow][acol0]     = *(u32x4*)&tmp[0];
        *(u32x4*)&As[arow][acol0 + 8] = *(u32x4*)&tmp[8];
        if (k0 + 32 < K) __builtin_prefetch((const float*)Ap + base + 32, 0, 1);
      }
    }
    // ---- stage B tile transposed (so B-frag reads are K-contiguous) ----
    {
      size_t base = (size_t)(k0 + brow) * N + bn * 128 + bcol0;
      const f32x4* p = (const f32x4*)(B + base);
#pragma unroll
      for (int i = 0; i < 4; i++) {
        f32x4 f = p[i];
        Bt[bcol0 + i * 4 + 0][brow] = f2bf(f.x);
        Bt[bcol0 + i * 4 + 1][brow] = f2bf(f.y);
        Bt[bcol0 + i * 4 + 2][brow] = f2bf(f.z);
        Bt[bcol0 + i * 4 + 3][brow] = f2bf(f.w);
      }
      if (k0 + 32 < K)
        __builtin_prefetch(B + base + (size_t)32 * N, 0, 1);
    }
    if constexpr (A_BF16)
      asm volatile("s_wait_asynccnt 0x0" ::: "memory");
    __syncthreads();

    // ---- fragments + 8 WMMAs per wave ----
    ABfrag aF[2], bF[4];
#pragma unroll
    for (int mt = 0; mt < 2; mt++) {
      int r = wm * 32 + mt * 16 + lrow;
      aF[mt].u[0] = *(const u32x4*)&As[r][koffA];
      aF[mt].u[1] = *(const u32x4*)&As[r][koffA + 8];
    }
#pragma unroll
    for (int nt = 0; nt < 4; nt++) {
      int c = wn * 64 + nt * 16 + lrow;
      bF[nt].u[0] = *(const u32x4*)&Bt[c][koffB];
      bF[nt].u[1] = *(const u32x4*)&Bt[c][koffB + 8];
    }
#pragma unroll
    for (int mt = 0; mt < 2; mt++)
#pragma unroll
      for (int nt = 0; nt < 4; nt++)
        acc[mt][nt] = WMMA_BF16(aF[mt].v, bF[nt].v, acc[mt][nt]);
    __syncthreads();
  }

  // ---- store C (D-layout: VGPR r -> row lhi*8+r, lane&15 -> col) ----
#pragma unroll
  for (int mt = 0; mt < 2; mt++) {
#pragma unroll
    for (int nt = 0; nt < 4; nt++) {
      int col = bn * 128 + wn * 64 + nt * 16 + lrow;
      float bv = 0.0f;
      if constexpr (OUT_F32) bv = bias ? bias[col] : 0.0f;
#pragma unroll
      for (int r = 0; r < 8; r++) {
        int row = bm * 128 + wm * 32 + mt * 16 + lhi * 8 + r;
        size_t idx = (size_t)row * N + col;
        if constexpr (OUT_F32)
          ((float*)Cp)[idx] = acc[mt][nt][r] + bv;
        else
          ((unsigned short*)Cp)[idx] = f2bf(acc[mt][nt][r]);
      }
    }
  }
}

// ---------------------------------------------------------------------------
// Flash attention: per block (qblock of 64 rows, head h, batch b).
// 128 threads = 4 waves; each wave owns 16 query rows.  Keys in blocks of 32
// (one WMMA K for the PV product), online softmax in fp32.
// q/k/v are bf16 [rows,1024] with head offset h*64.
// ---------------------------------------------------------------------------
__global__ __launch_bounds__(128) void attn_fwd(
    const unsigned short* __restrict__ q, const unsigned short* __restrict__ k,
    const unsigned short* __restrict__ v, unsigned short* __restrict__ o,
    int N, int Mk) {
  __shared__ alignas(16) unsigned short Ks[32][64];     // [key][d]
  __shared__ alignas(16) unsigned short Vt[64][32];     // [d][key] (transposed)
  __shared__ alignas(16) unsigned short Ps[4][16][32];  // per-wave P tile

  const int tid  = threadIdx.x;
  const int wave = tid >> 5, lane = tid & 31;
  const int lrow = lane & 15, lhi = lane >> 4;
  const int koffA = lhi * 16, koffB = lhi * 16;
  const int b = blockIdx.z, h = blockIdx.y;
  const int qrow0 = b * N + blockIdx.x * 64 + wave * 16;
  const int hc = h * 64;

  // Q A-fragments for the two 32-wide d chunks (live in registers all loop)
  ABfrag qA[2];
#pragma unroll
  for (int c = 0; c < 2; c++) {
    const unsigned short* p = q + (size_t)(qrow0 + lrow) * 1024 + hc + c * 32;
    qA[c].u[0] = *(const u32x4*)(p + koffA);
    qA[c].u[1] = *(const u32x4*)(p + koffA + 8);
  }

  float mrow[8], lsum[8];
#pragma unroll
  for (int r = 0; r < 8; r++) { mrow[r] = -1e30f; lsum[r] = 0.0f; }
  v8f accO[4];
  {
    v8f z = {};
#pragma unroll
    for (int t = 0; t < 4; t++) accO[t] = z;
  }

  const int srow = tid >> 2, sseg = (tid & 3) * 16;  // staging: 32 rows x 4 segs

  for (int jb = 0; jb < Mk / 32; jb++) {
    const int kb0 = jb * 32;
    // ---- stage K block [32 keys][64 d]: async DMA memory -> LDS ----
    {
      unsigned lds = (unsigned)(uintptr_t)&Ks[srow][sseg];
      const unsigned short* gp =
          k + (size_t)(b * Mk + kb0 + srow) * 1024 + hc + sseg;
      asm volatile("global_load_async_to_lds_b128 %0, %1, off" ::"v"(lds),
                   "v"(gp)
                   : "memory");
      asm volatile(
          "global_load_async_to_lds_b128 %0, %1, off offset:16" ::"v"(lds),
          "v"(gp)
          : "memory");
    }
    // ---- stage V block transposed -> Vt[d][key] (VALU transpose) ----
    {
      const unsigned short* p =
          v + (size_t)(b * Mk + kb0 + srow) * 1024 + hc + sseg;
      alignas(16) unsigned short tmp[16];
      *(u32x4*)&tmp[0] = *(const u32x4*)p;
      *(u32x4*)&tmp[8] = *(const u32x4*)(p + 8);
#pragma unroll
      for (int j = 0; j < 16; j++) Vt[sseg + j][srow] = tmp[j];
    }
    // prefetch next key block
    if (jb + 1 < Mk / 32) {
      __builtin_prefetch(
          k + (size_t)(b * Mk + kb0 + 32 + srow) * 1024 + hc + sseg, 0, 1);
      __builtin_prefetch(
          v + (size_t)(b * Mk + kb0 + 32 + srow) * 1024 + hc + sseg, 0, 1);
    }
    asm volatile("s_wait_asynccnt 0x0" ::: "memory");
    __syncthreads();

    // ---- S = q @ K^T (two 16-key tiles, two d chunks each) ----
    v8f S[2];
#pragma unroll
    for (int t = 0; t < 2; t++) {
      const unsigned short* kr = &Ks[t * 16 + lrow][0];
      ABfrag kB0, kB1;
      kB0.u[0] = *(const u32x4*)(kr + koffB);
      kB0.u[1] = *(const u32x4*)(kr + koffB + 8);
      kB1.u[0] = *(const u32x4*)(kr + 32 + koffB);
      kB1.u[1] = *(const u32x4*)(kr + 32 + koffB + 8);
      v8f s = {};
      s = WMMA_BF16(qA[0].v, kB0.v, s);
      s = WMMA_BF16(qA[1].v, kB1.v, s);
      S[t] = s;
    }

    // ---- online softmax (row = lhi*8+r; cols across the 16-lane row) ----
    const float scale = 0.125f;  // 64^-0.5
#pragma unroll
    for (int r = 0; r < 8; r++) {
      float s0 = S[0][r] * scale, s1 = S[1][r] * scale;
      float m = rowmax16(fmaxf(s0, s1));
      float mnew = fmaxf(mrow[r], m);
      float corr = __expf(mrow[r] - mnew);
      float p0 = __expf(s0 - mnew), p1 = __expf(s1 - mnew);
      float rs = rowsum16(p0 + p1);
      lsum[r] = lsum[r] * corr + rs;
      mrow[r] = mnew;
#pragma unroll
      for (int t = 0; t < 4; t++) accO[t][r] *= corr;
      // D-layout -> A-layout round trip through per-wave LDS tile
      Ps[wave][lhi * 8 + r][lrow]      = f2bf(p0);
      Ps[wave][lhi * 8 + r][16 + lrow] = f2bf(p1);
    }
    // same-wave LDS RAW: make the P stores visible before fragment reads
    asm volatile("s_wait_dscnt 0x0" ::: "memory");

    ABfrag pA;
    {
      const unsigned short* pr = &Ps[wave][lrow][0];
      pA.u[0] = *(const u32x4*)(pr + koffA);
      pA.u[1] = *(const u32x4*)(pr + koffA + 8);
    }
    // ---- O += P @ V (four 16-wide d tiles, K=32 keys) ----
#pragma unroll
    for (int t = 0; t < 4; t++) {
      const unsigned short* vr = &Vt[t * 16 + lrow][0];
      ABfrag vB;
      vB.u[0] = *(const u32x4*)(vr + koffB);
      vB.u[1] = *(const u32x4*)(vr + koffB + 8);
      accO[t] = WMMA_BF16(pA.v, vB.v, accO[t]);
    }
    __syncthreads();
  }

  // ---- normalize and store bf16 attention output ----
#pragma unroll
  for (int t = 0; t < 4; t++) {
#pragma unroll
    for (int r = 0; r < 8; r++) {
      float outv = accO[t][r] / lsum[r];
      size_t idx = (size_t)(qrow0 + lhi * 8 + r) * 1024 + hc + t * 16 + lrow;
      o[idx] = f2bf(outv);
    }
  }
}

// ---------------------------------------------------------------------------
extern "C" void kernel_launch(void* const* d_in, const int* in_sizes, int n_in,
                              void* d_out, int out_size, void* d_ws,
                              size_t ws_size, hipStream_t stream) {
  const float* x   = (const float*)d_in[0];
  const float* ctx = (const float*)d_in[1];
  const float* Wq  = (const float*)d_in[2];
  const float* Wk  = (const float*)d_in[3];
  const float* Wv  = (const float*)d_in[4];
  const float* Wo  = (const float*)d_in[5];
  const float* bo  = (const float*)d_in[6];

  const int B = 4, N = 4096, Mk = 1024, INNER = 1024, QD = 1024, CD = 768;

  // bf16 workspace: q (32MB) | k (8MB) | v (8MB) | attn_out (32MB) = 80MB
  unsigned short* qb = (unsigned short*)d_ws;
  unsigned short* kb = qb + (size_t)B * N * INNER;
  unsigned short* vb = kb + (size_t)B * Mk * INNER;
  unsigned short* ab = vb + (size_t)B * Mk * INNER;

  // q = x @ Wq  (fp32 in, bf16 out)
  gemm_bf16_wmma<false, false>
      <<<dim3(INNER / 128, (B * N) / 128), 256, 0, stream>>>(
          x, Wq, qb, nullptr, B * N, INNER, QD);
  // k = ctx @ Wk
  gemm_bf16_wmma<false, false>
      <<<dim3(INNER / 128, (B * Mk) / 128), 256, 0, stream>>>(
          ctx, Wk, kb, nullptr, B * Mk, INNER, CD);
  // v = ctx @ Wv
  gemm_bf16_wmma<false, false>
      <<<dim3(INNER / 128, (B * Mk) / 128), 256, 0, stream>>>(
          ctx, Wv, vb, nullptr, B * Mk, INNER, CD);
  // fused flash attention
  attn_fwd<<<dim3(N / 64, 16, B), 128, 0, stream>>>(qb, kb, vb, ab, N, Mk);
  // out = attn @ Wo + bo  (bf16 in, fp32 out)
  gemm_bf16_wmma<true, true>
      <<<dim3(QD / 128, (B * N) / 128), 256, 0, stream>>>(
          ab, Wo, d_out, bo, B * N, QD, INNER);
}
